// HINormer_24927990186124
// MI455X (gfx1250) — matrix operand order repaired and checked by reading, not compiled, not run on hardware
//
#include <hip/hip_runtime.h>
#include <hip/hip_bf16.h>
#include <stdint.h>

// ---------------------------------------------------------------------------
// Problem constants (match reference)
// ---------------------------------------------------------------------------
#define B_   4
#define S_   2048
#define D_   512
#define H_   8
#define HD_  64
#define RL_  64
#define MT_  (B_ * S_)        // 8192 rows total
#define SLOPE_ 0.01f
#define LN_EPS_ 1e-5f

typedef __bf16 bf16_t;
typedef __attribute__((ext_vector_type(16))) __bf16 v16bf;
typedef __attribute__((ext_vector_type(8)))  __bf16 v8bf;
typedef __attribute__((ext_vector_type(8)))  float  v8f;
typedef __attribute__((ext_vector_type(4)))  unsigned int u32x4;
typedef __attribute__((ext_vector_type(4)))  int i32x4;
typedef __attribute__((ext_vector_type(8)))  int i32x8;

#if defined(__gfx1250__) && __has_builtin(__builtin_amdgcn_tensor_load_to_lds) && \
    __has_builtin(__builtin_amdgcn_s_wait_tensorcnt)
#define USE_TDM 1
#endif

#define WMMA_BF16(A, Bm, C) \
    __builtin_amdgcn_wmma_f32_16x16x32_bf16(false, (A), false, (Bm), (short)0, (C), false, false)

static __device__ __forceinline__ v8f vzero8() {
    v8f z = {0.f, 0.f, 0.f, 0.f, 0.f, 0.f, 0.f, 0.f};
    return z;
}

// 16 contiguous bf16 (two 16B chunks). Requires 16B alignment.
static __device__ __forceinline__ v16bf ld16(const bf16_t* p) {
    union { v16bf v; v8bf h[2]; } u;
    u.h[0] = *(const v8bf*)(p);
    u.h[1] = *(const v8bf*)(p + 8);
    return u.v;
}
// A-fragment chunks: [p..p+7] and [p+16..p+23]  (16-bit A 16x32 layout:
// lane<16 holds K {kb..kb+7, kb+16..kb+23}, kb = 8*(lane>>4)).
static __device__ __forceinline__ v16bf ld16gap(const bf16_t* p) {
    union { v16bf v; v8bf h[2]; } u;
    u.h[0] = *(const v8bf*)(p);
    u.h[1] = *(const v8bf*)(p + 16);
    return u.v;
}

static __device__ __forceinline__ v16bf ones16() {
    union { v16bf v; bf16_t b[16]; } u;
#pragma unroll
    for (int i = 0; i < 16; ++i) u.b[i] = (bf16_t)1.0f;
    return u.v;
}

// ---------------------------------------------------------------------------
// Generic GEMM:  C[M x N] = A[M x K] * W[K x N]
//   A, W fp32 row-major; output type OT (bf16 or fp32), branch-free epilogue.
//   Block = 256 threads = 8 waves arranged 4(M) x 2(N); block tile 128x128,
//   wave tile 32x64 (2x4 of 16x16 WMMA -> 8 WMMAs / K-step), K step 32.
// ---------------------------------------------------------------------------
#define GBM 128
#define GBN 128
#define GBK 32
#define LDA_S 40   // A tile LDS row stride (bf16): 80B rows, 8B/16B aligned
#define LDW_S 48   // W^T tile LDS row stride: 96B rows, 16B aligned

template <typename OT>
__global__ __launch_bounds__(256)
void hin_gemm(const float* __restrict__ A, const float* __restrict__ W,
              OT* __restrict__ out, int M, int N, int K)
{
    __shared__ __align__(32) bf16_t At[GBM * LDA_S];
    __shared__ __align__(32) bf16_t Wt[GBN * LDW_S];

    const int tid  = threadIdx.x;
    const int wave = tid >> 5;
    const int lane = tid & 31;
    const int wm   = wave >> 1;        // 0..3  (32-row slice)
    const int wn   = wave & 1;         // 0..1  (64-col slice)
    const int l15  = lane & 15;
    const int lhi  = lane >> 4;
    const int m0   = blockIdx.x * GBM;
    const int n0   = blockIdx.y * GBN;

    v8f acc[2][4];
    for (int i = 0; i < 2; ++i)
        for (int j = 0; j < 4; ++j) acc[i][j] = vzero8();

    for (int k0 = 0; k0 < K; k0 += GBK) {
        __syncthreads();
        // Prefetch next A tile rows (emits global_prefetch_b8).
        if (k0 + GBK < K && tid < GBM)
            __builtin_prefetch(&A[(size_t)(m0 + tid) * K + k0 + GBK], 0, 1);
        // Stage A tile 128x32: float4 loads, packed bf16x4 (b64) LDS stores.
        for (int idx = tid; idx < GBM * (GBK / 4); idx += 256) {
            int r = idx >> 3, c4 = (idx & 7) << 2;
            float4 v = *(const float4*)&A[(size_t)(m0 + r) * K + (k0 + c4)];
            union { bf16_t b[4]; uint64_t u; } pk;
            pk.b[0] = (bf16_t)v.x; pk.b[1] = (bf16_t)v.y;
            pk.b[2] = (bf16_t)v.z; pk.b[3] = (bf16_t)v.w;
            *(uint64_t*)&At[r * LDA_S + c4] = pk.u;
        }
        // Stage W tile 32x128 transposed (Wt[n][k]): float4 loads, b16 scatter.
        for (int idx = tid; idx < GBK * (GBN / 4); idx += 256) {
            int r = idx >> 5, c4 = (idx & 31) << 2;   // r = k, c4 = n
            float4 v = *(const float4*)&W[(size_t)(k0 + r) * N + (n0 + c4)];
            Wt[(c4 + 0) * LDW_S + r] = (bf16_t)v.x;
            Wt[(c4 + 1) * LDW_S + r] = (bf16_t)v.y;
            Wt[(c4 + 2) * LDW_S + r] = (bf16_t)v.z;
            Wt[(c4 + 3) * LDW_S + r] = (bf16_t)v.w;
        }
        __syncthreads();

        v16bf af[2], bw[4];
        const int kb = lhi * 8;
        for (int i = 0; i < 2; ++i)
            af[i] = ld16gap(&At[(wm * 32 + i * 16 + l15) * LDA_S + kb]);
        for (int j = 0; j < 4; ++j)
            bw[j] = ld16gap(&Wt[(wn * 64 + j * 16 + l15) * LDW_S + lhi * 8]);

        for (int i = 0; i < 2; ++i)
            for (int j = 0; j < 4; ++j)
                acc[i][j] = WMMA_BF16(af[i], bw[j], acc[i][j]);
    }

    // Branch-free epilogue: C element r -> (M = r + 8*lhi, N = l15).
    const int rowb = m0 + wm * 32 + 8 * lhi;
    const int colb = n0 + wn * 64 + l15;
    for (int i = 0; i < 2; ++i)
        for (int j = 0; j < 4; ++j)
            for (int r = 0; r < 8; ++r) {
                int row = rowb + i * 16 + r;
                int col = colb + j * 16;
                out[(size_t)row * N + col] = (OT)acc[i][j][r];
            }
}

// ---------------------------------------------------------------------------
// Additive attention logits: s[row, h] = sum_d leaky(F[row, h*64+d]) * a[d]
// ---------------------------------------------------------------------------
__global__ __launch_bounds__(256)
void hin_head_score(const bf16_t* __restrict__ F, const float* __restrict__ avec,
                    float* __restrict__ out)
{
    int idx = blockIdx.x * 256 + threadIdx.x;     // MT_*H_ threads
    if (idx >= MT_ * H_) return;
    int row = idx >> 3, h = idx & 7;
    const v8bf* p8 = (const v8bf*)(F + (size_t)row * D_ + h * HD_);
    float acc = 0.f;
    for (int c = 0; c < 8; ++c) {
        v8bf v = p8[c];
#pragma unroll
        for (int d = 0; d < 8; ++d) {
            float x = (float)v[d];
            x = (x >= 0.f) ? x : SLOPE_ * x;
            acc += x * avec[c * 8 + d];
        }
    }
    out[idx] = acc;
}

// ---------------------------------------------------------------------------
// Fused attention (flash style), per (b,h):
//   logits = sl[i] + sr[j] + r_k[i,:] . r_q[j,:]   (K = 64, bf16 WMMA)
//   online softmax over j; context += P @ fr       (bf16 WMMA)
//   row-sums of P via an extra ones-matrix WMMA (no cross-lane sum shuffles).
// Block: 256 thr = 8 waves; wave owns 16 query rows; 64 keys / iteration.
// r_q tile staged by the Tensor Data Mover (2-D tile + LDS row padding);
// fr tile staged cooperatively (needs a transpose TDM can't do).
// ---------------------------------------------------------------------------
#define KT   64
#define RQ_S 80   // [64 keys][64] stride  -> 160B rows (128B data + 32B pad)
#define FR_S 72   // frT [64 d][64 keys]   -> 144B rows, 16B aligned
#define P_S  72   // P [16 rows][64 keys]  -> 144B rows, 16B aligned

struct AttnSmem {
    bf16_t RQ[KT * RQ_S];        // at LDS offset 0 (TDM descriptor assumes this)
    bf16_t FT[HD_ * FR_S];
    bf16_t PT[8][16 * P_S];
};

__global__ __launch_bounds__(256)
void hin_attn(const bf16_t* __restrict__ rk, const bf16_t* __restrict__ rq,
              const bf16_t* __restrict__ fr,
              const float* __restrict__ sl, const float* __restrict__ sr,
              float* __restrict__ hsa)
{
    __shared__ __align__(32) AttnSmem sm;

    const int tid  = threadIdx.x;
    const int wave = tid >> 5;
    const int lane = tid & 31;
    const int l15  = lane & 15;
    const int lhi  = lane >> 4;
    const int bh   = blockIdx.y;
    const int b    = bh >> 3, h = bh & 7;
    const int qs   = blockIdx.x * 128 + wave * 16;     // query start within S
    const size_t rowbase = (size_t)b * S_;

    // Preload r_k A-fragments for this wave's 16 query rows (K = 64 -> 2 frags).
    v16bf akf[2];
    {
        const bf16_t* pa = rk + (rowbase + qs + l15) * D_ + h * HD_;
        for (int ks = 0; ks < 2; ++ks)
            akf[ks] = ld16gap(pa + ks * 32 + lhi * 8);
    }
    // Row logits sl (row M = r + 8*lhi within the tile).
    float slr[8];
    for (int r = 0; r < 8; ++r)
        slr[r] = sl[(rowbase + qs + r + 8 * lhi) * H_ + h];

    const v16bf onesf = ones16();
    float m_[8];
    for (int r = 0; r < 8; ++r) m_[r] = -1e30f;
    v8f o[4], osum;
    for (int n = 0; n < 4; ++n) o[n] = vzero8();
    osum = vzero8();

    bf16_t* pw = &sm.PT[wave][0];

    for (int j0 = 0; j0 < S_; j0 += KT) {
        __syncthreads();

#ifdef USE_TDM
        // ---- TDM: DMA r_q tile [64 rows x 64 bf16] into padded LDS rows ----
        if (wave == 0) {
            uint64_t gaddr = (uint64_t)(uintptr_t)(rq + (rowbase + j0) * D_ + h * HD_);
            u32x4 g0;
            g0[0] = 1u;                                     // count=1, user mode
            g0[1] = 0u;                                     // lds_addr = offset of sm.RQ
            g0[2] = (unsigned)gaddr;                        // global_addr[31:0]
            g0[3] = (unsigned)((gaddr >> 32) & 0x01FFFFFFu) // global_addr[56:32]
                  | 0x80000000u;                            // type = 2 ("image")
            i32x8 g1;
            g1[0] = (1 << 16)     // data_size = 2 bytes
                  | (1 << 20)     // pad_enable
                  | (4 << 22)     // pad_interval: 32 DWORDs (128B row)
                  | (7 << 25);    // pad_amount:   8 DWORDs (32B -> 160B stride)
            g1[1] = (int)(512u << 16);                      // tensor_dim0 = 512
            g1[2] = (int)(((unsigned)MT_ & 0xFFFFu) << 16); // tensor_dim1 = 8192
            g1[3] = (int)(64u << 16);                       // tile_dim0 = 64
            g1[4] = KT;                                     // tile_dim1 = 64 rows
            g1[5] = 512;                                    // tensor_dim0_stride
            g1[6] = 0;
            g1[7] = 0;
            i32x4 z4 = {0, 0, 0, 0};
#if __clang_major__ >= 23
            i32x8 z8 = {0, 0, 0, 0, 0, 0, 0, 0};
            __builtin_amdgcn_tensor_load_to_lds(g0, g1, z4, z4, z8, 0);
#else
            __builtin_amdgcn_tensor_load_to_lds(g0, g1, z4, z4, 0);
#endif
        }
#else
        for (int idx = tid; idx < KT * 8; idx += 256) {
            int kr = idx >> 3, c8 = (idx & 7) << 3;
            *(v8bf*)&sm.RQ[kr * RQ_S + c8] =
                *(const v8bf*)&rq[(rowbase + j0 + kr) * D_ + h * HD_ + c8];
        }
#endif
        // ---- cooperative transpose-stage of fr tile: FT[d][key] ----
        for (int idx = tid; idx < KT * 8; idx += 256) {
            int kr = idx >> 3, c8 = (idx & 7) << 3;
            v8bf v = *(const v8bf*)&fr[(rowbase + j0 + kr) * D_ + h * HD_ + c8];
#pragma unroll
            for (int d = 0; d < 8; ++d)
                sm.FT[(c8 + d) * FR_S + kr] = v[d];
        }
#ifdef USE_TDM
        if (wave == 0) __builtin_amdgcn_s_wait_tensorcnt(0);
#endif
        __syncthreads();

        // ---- score: four 16-key subtiles, each K=64 via 2 WMMAs ----
        v8f sc[4];
        for (int jt = 0; jt < 4; ++jt) {
            v8f c = vzero8();
            for (int ks = 0; ks < 2; ++ks) {
                v16bf bq = ld16gap(&sm.RQ[(jt * 16 + l15) * RQ_S + ks * 32 + lhi * 8]);
                c = WMMA_BF16(akf[ks], bq, c);
            }
            float srv = sr[(rowbase + j0 + jt * 16 + l15) * H_ + h];
            for (int r = 0; r < 8; ++r) c[r] = c[r] + slr[r] + srv;
            sc[jt] = c;
        }

        // ---- online softmax over the 64 keys (max via 16-lane butterfly) ----
        float mnew[8], alpha[8];
        for (int r = 0; r < 8; ++r) {
            float x = fmaxf(fmaxf(sc[0][r], sc[1][r]), fmaxf(sc[2][r], sc[3][r]));
            for (int msk = 8; msk >= 1; msk >>= 1)
                x = fmaxf(x, __shfl_xor(x, msk, 32));
            mnew[r] = fmaxf(m_[r], x);
            alpha[r] = __expf(m_[r] - mnew[r]);
            m_[r] = mnew[r];
        }
        for (int jt = 0; jt < 4; ++jt)
            for (int r = 0; r < 8; ++r) {
                float p = __expf(sc[jt][r] - mnew[r]);
                // C-layout -> row-major P patch in LDS (per-wave region).
                pw[(r + 8 * lhi) * P_S + jt * 16 + l15] = (bf16_t)p;
            }
        for (int n = 0; n < 4; ++n)
            for (int r = 0; r < 8; ++r) o[n][r] *= alpha[r];
        for (int r = 0; r < 8; ++r) osum[r] *= alpha[r];

        // ---- context: P[16x64] (A, 2 K-steps) @ fr[64x64] (B, 4 N-subtiles),
        //      plus P @ ones for the softmax denominators.                 ----
        for (int ks = 0; ks < 2; ++ks) {
            v16bf pf = ld16gap(&pw[l15 * P_S + ks * 32 + lhi * 8]);
            for (int n = 0; n < 4; ++n) {
                v16bf bv = ld16gap(&sm.FT[(n * 16 + l15) * FR_S + ks * 32 + lhi * 8]);
                o[n] = WMMA_BF16(pf, bv, o[n]);
            }
            osum = WMMA_BF16(pf, onesf, osum);
        }
    }

    // Normalize rows (osum already broadcast across lanes) and scatter to
    // h_sa [B*S, 512] (col = h*64 + d).
    float inv[8];
    for (int r = 0; r < 8; ++r) inv[r] = 1.f / osum[r];
    for (int n = 0; n < 4; ++n)
        for (int r = 0; r < 8; ++r) {
            size_t row = rowbase + qs + r + 8 * lhi;
            hsa[row * D_ + h * HD_ + n * 16 + l15] = o[n][r] * inv[r];
        }
}

// ---------------------------------------------------------------------------
// out = LayerNorm(h + fh) * g + b       (one 256-thread block per row)
// ---------------------------------------------------------------------------
__global__ __launch_bounds__(256)
void hin_layernorm(const float* __restrict__ hin, const float* __restrict__ fh,
                   const float* __restrict__ gam, const float* __restrict__ bet,
                   float* __restrict__ out)
{
    const int row = blockIdx.x;
    const int tid = threadIdx.x;
    const float* ph = hin + (size_t)row * D_;
    const float* pf = fh  + (size_t)row * D_;
    __shared__ float sbuf[8];
    __shared__ float smu, srs;

    float x0 = ph[tid]       + pf[tid];
    float x1 = ph[tid + 256] + pf[tid + 256];
    float s = x0 + x1;
    for (int m = 16; m >= 1; m >>= 1) s += __shfl_xor(s, m, 32);
    if ((tid & 31) == 0) sbuf[tid >> 5] = s;
    __syncthreads();
    if (tid == 0) {
        float t = 0.f;
        for (int i = 0; i < 8; ++i) t += sbuf[i];
        smu = t / (float)D_;
    }
    __syncthreads();
    float mu = smu;
    float d0 = x0 - mu, d1 = x1 - mu;
    s = d0 * d0 + d1 * d1;
    for (int m = 16; m >= 1; m >>= 1) s += __shfl_xor(s, m, 32);
    if ((tid & 31) == 0) sbuf[tid >> 5] = s;
    __syncthreads();
    if (tid == 0) {
        float t = 0.f;
        for (int i = 0; i < 8; ++i) t += sbuf[i];
        srs = rsqrtf(t / (float)D_ + LN_EPS_);
    }
    __syncthreads();
    float rs = srs;
    out[(size_t)row * D_ + tid]       = d0 * rs * gam[tid]       + bet[tid];
    out[(size_t)row * D_ + tid + 256] = d1 * rs * gam[tid + 256] + bet[tid + 256];
}

// ---------------------------------------------------------------------------
// Host-side orchestration
// ---------------------------------------------------------------------------
extern "C" void kernel_launch(void* const* d_in, const int* in_sizes, int n_in,
                              void* d_out, int out_size, void* d_ws, size_t ws_size,
                              hipStream_t stream)
{
    (void)in_sizes; (void)n_in; (void)out_size; (void)ws_size;
    const float* h   = (const float*)d_in[0];
    const float* rh  = (const float*)d_in[1];
    const float* Wl  = (const float*)d_in[2];
    const float* Wr  = (const float*)d_in[3];
    const float* al  = (const float*)d_in[4];
    const float* ar  = (const float*)d_in[5];
    const float* Wrs = (const float*)d_in[6];
    const float* Wrt = (const float*)d_in[7];
    const float* Wf  = (const float*)d_in[8];
    const float* lng = (const float*)d_in[9];
    const float* lnb = (const float*)d_in[10];
    float* out = (float*)d_out;

    uint8_t* wsp = (uint8_t*)d_ws;
    size_t off = 0;
    auto take = [&](size_t bytes) -> void* {
        void* p = wsp + off;
        off += (bytes + 255) & ~(size_t)255;
        return p;
    };
    bf16_t* fl  = (bf16_t*)take((size_t)MT_ * D_ * sizeof(bf16_t));
    bf16_t* fr  = (bf16_t*)take((size_t)MT_ * D_ * sizeof(bf16_t));
    bf16_t* rk  = (bf16_t*)take((size_t)MT_ * D_ * sizeof(bf16_t));
    bf16_t* rq  = (bf16_t*)take((size_t)MT_ * D_ * sizeof(bf16_t));
    float*  slb = (float*) take((size_t)MT_ * H_ * sizeof(float));
    float*  srb = (float*) take((size_t)MT_ * H_ * sizeof(float));
    float*  hsa = (float*) take((size_t)MT_ * D_ * sizeof(float));
    float*  fh  = (float*) take((size_t)MT_ * D_ * sizeof(float));

    dim3 gGemm(MT_ / GBM, D_ / GBN);   // 64 x 4
    // 1) projections (bf16 outputs)
    hin_gemm<bf16_t><<<gGemm, 256, 0, stream>>>(h,  Wl,  fl, MT_, D_, D_);
    hin_gemm<bf16_t><<<gGemm, 256, 0, stream>>>(h,  Wr,  fr, MT_, D_, D_);
    // 2) relation embeddings (K = 64)
    hin_gemm<bf16_t><<<gGemm, 256, 0, stream>>>(rh, Wrs, rk, MT_, D_, RL_);
    hin_gemm<bf16_t><<<gGemm, 256, 0, stream>>>(rh, Wrt, rq, MT_, D_, RL_);
    // 3) additive logits
    hin_head_score<<<(MT_ * H_ + 255) / 256, 256, 0, stream>>>(fl, al, slb);
    hin_head_score<<<(MT_ * H_ + 255) / 256, 256, 0, stream>>>(fr, ar, srb);
    // 4) fused flash attention (TDM-staged r_q tiles)
    dim3 gAttn(S_ / 128, B_ * H_);     // 16 x 32
    hin_attn<<<gAttn, 256, 0, stream>>>(rk, rq, fr, slb, srb, hsa);
    // 5) output projection (fp32 output)
    hin_gemm<float><<<gGemm, 256, 0, stream>>>(hsa, Wf, fh, MT_, D_, D_);
    // 6) residual + layernorm
    hin_layernorm<<<MT_, 256, 0, stream>>>(h, fh, lng, lnb, out);
}
